// LocalAttention_28063316313032
// MI455X (gfx1250) — matrix operand order repaired and checked by reading, not compiled, not run on hardware
//
#include <hip/hip_runtime.h>
#include <hip/hip_bf16.h>
#include <stdint.h>

// ---------------------------------------------------------------------------
// Swin window attention for MI455X (gfx1250, wave32, WMMA bf16 16x16x32)
//   B=4096 windows, N=49 tokens (padded to 64), C=384, H=12 heads, hd=32
// Pipeline: [tables] -> [QKV gemm] -> [fused attention] -> [out gemm]
// Matmuls on v_wmma_f32_16x16x32_bf16; LDS staging via TDM tensor_load_to_lds
// where data is already bf16 (attention Q/K/V tiles, out-proj ctx tile).
// ---------------------------------------------------------------------------

typedef __attribute__((ext_vector_type(16))) __bf16 bf16x16;
typedef __attribute__((ext_vector_type(8)))  float  f32x8;
typedef unsigned int u32x4 __attribute__((ext_vector_type(4)));
typedef int          i32x8 __attribute__((ext_vector_type(8)));
typedef int          i32x4 __attribute__((ext_vector_type(4)));

#define HEADS 12
#define NTOK  49
#define NPAD  64
#define CDIM  384
#define HDIM  32
#define NWIN  64
#define BWIN  4096

#if defined(__gfx1250__) && __has_builtin(__builtin_amdgcn_tensor_load_to_lds) && \
    __has_builtin(__builtin_amdgcn_s_wait_tensorcnt)
#define USE_TDM 1
#else
#define USE_TDM 0
#endif

static __device__ __forceinline__ f32x8 wmma_bf16(bf16x16 a, bf16x16 b, f32x8 c) {
  return __builtin_amdgcn_wmma_f32_16x16x32_bf16(false, a, false, b, (short)0, c,
                                                 false, false);
}

union Pack8  { uint4 u; __bf16 h[8]; };
union Pack4h { uint2 u; __bf16 h[4]; };

static __device__ __forceinline__ f32x8 zero8() {
  f32x8 z = {0.f, 0.f, 0.f, 0.f, 0.f, 0.f, 0.f, 0.f};
  return z;
}

#if USE_TDM
// Tensor Data Mover: DMA a 2D tile of bf16 (data_size=2B) from global to LDS.
// D# layout per CDNA5 ISA:
//  group0: [1:0]=count=1, [63:32]=lds_addr, [120:64]=global_addr, [127:126]=type=2
//  group1: [17:16]=data_size(1->2B), [79:48]=tensor_dim0, [111:80]=tensor_dim1,
//          [127:112]=tile_dim0, [143:128]=tile_dim1, [207:160]=tensor_dim0_stride
// This toolchain exposes the 6-arg builtin:
//   (uint32x4 g0, int32x8 g1, int32x4 g2, int32x4 g3, int32x8 pad, int cpol)
static __device__ __forceinline__ void tdm_load_bf16_2d(const void* gptr,
                                                        unsigned lds_off,
                                                        unsigned rows,
                                                        unsigned rowElems,
                                                        unsigned strideElems) {
  unsigned long long ga = (unsigned long long)(uintptr_t)gptr;
  u32x4 g0;
  g0[0] = 1u;                                           // count=1 (valid user D#)
  g0[1] = lds_off;                                      // LDS byte address
  g0[2] = (unsigned)(ga & 0xFFFFFFFFu);                 // global_addr[31:0]
  g0[3] = (unsigned)((ga >> 32) & 0x01FFFFFFu) | (2u << 30);  // [56:32] | type=2
  i32x8 g1;
  g1[0] = (int)(1u << 16);                              // wg_mask=0, data_size=2B
  g1[1] = (int)((rowElems & 0xFFFFu) << 16);            // tensor_dim0[15:0]
  g1[2] = (int)(((rowElems >> 16) & 0xFFFFu) |
                ((rows & 0xFFFFu) << 16));              // td0[31:16] | td1[15:0]
  g1[3] = (int)(((rows >> 16) & 0xFFFFu) |
                ((rowElems & 0xFFFFu) << 16));          // td1[31:16] | tile_dim0
  g1[4] = (int)(rows & 0xFFFFu);                        // tile_dim1 | tile_dim2=0
  g1[5] = (int)strideElems;                             // tensor_dim0_stride[31:0]
  g1[6] = 0;                                            // stride0[47:32]|stride1 lo
  g1[7] = 0;
  i32x4 z4v = {0, 0, 0, 0};
  i32x8 z8v = {0, 0, 0, 0, 0, 0, 0, 0};
  __builtin_amdgcn_tensor_load_to_lds(g0, g1, z4v, z4v, z8v, 0);
}

static __device__ __forceinline__ unsigned lds_offset_of(const void* p) {
  // generic pointer to LDS: addr[31:0] is the LDS byte offset (ISA §10.2)
  return (unsigned)(uintptr_t)p;
}
#endif

// ---------------------------------------------------------------------------
// Kernel D: padded bias table [12][64][64] and mask table [64][64][64].
// Padding columns >=49 get -1e30 so softmax zeroes them.
// ---------------------------------------------------------------------------
__global__ __launch_bounds__(256)
void build_tables(const float* __restrict__ mask,
                  const float* __restrict__ rpb,
                  const int*   __restrict__ relidx,
                  float* __restrict__ biasP,
                  float* __restrict__ maskP) {
  int idx = blockIdx.x * blockDim.x + threadIdx.x;
  if (idx < HEADS * NPAD * NPAD) {
    int h = idx / (NPAD * NPAD);
    int rem = idx % (NPAD * NPAD);
    int m = rem / NPAD, n = rem % NPAD;
    float v = 0.f;
    if (m < NTOK && n < NTOK) v = rpb[relidx[m * NTOK + n] * HEADS + h];
    biasP[idx] = v;
  }
  int idx2 = idx - HEADS * NPAD * NPAD;
  if (idx2 >= 0 && idx2 < NWIN * NPAD * NPAD) {
    int w = idx2 / (NPAD * NPAD);
    int rem = idx2 % (NPAD * NPAD);
    int m = rem / NPAD, n = rem % NPAD;
    float v;
    if (n >= NTOK)      v = -1e30f;
    else if (m >= NTOK) v = 0.f;
    else                v = mask[(w * NTOK + m) * NTOK + n];
    maskP[idx2] = v;
  }
}

// ---------------------------------------------------------------------------
// Kernel A: qkv = x @ W_qkv + b_qkv (M=49 pad 64, N=1152, K=384)
// x is f32 -> must convert to bf16, so staging stays on the VALU path here.
// ---------------------------------------------------------------------------
__global__ __launch_bounds__(256)
void qkv_gemm(const float* __restrict__ x,
              const float* __restrict__ Wqkv,
              const float* __restrict__ bqkv,
              __bf16* __restrict__ Qp,
              __bf16* __restrict__ Kp,
              __bf16* __restrict__ Vp) {
  __shared__ __bf16 xs[NPAD * CDIM];        // 48 KB
  const int b    = blockIdx.y;
  const int cb   = blockIdx.x;
  const int tid  = threadIdx.x;
  const int wave = tid >> 5, lane = tid & 31;

  const float* xrow = x + (size_t)b * NTOK * CDIM;
  for (int idx = tid; idx < NPAD * (CDIM / 4); idx += 256) {
    int row = idx / (CDIM / 4);
    int c4  = idx % (CDIM / 4);
    float4 v = make_float4(0.f, 0.f, 0.f, 0.f);
    if (row < NTOK) v = *(const float4*)(xrow + row * CDIM + c4 * 4);
    Pack4h p;
    p.h[0] = (__bf16)v.x; p.h[1] = (__bf16)v.y;
    p.h[2] = (__bf16)v.z; p.h[3] = (__bf16)v.w;
    *(uint2*)&xs[row * CDIM + c4 * 4] = p.u;
  }
  __syncthreads();

  const int colBase = cb * 128 + wave * 16;
  const int col     = colBase + (lane & 15);
  const int kb      = (lane < 16) ? 0 : 16;
  const int ah      = (lane < 16) ? 0 : 16;
  const int arowl   = lane & 15;

  f32x8 acc[4] = {zero8(), zero8(), zero8(), zero8()};
  for (int kk = 0; kk < CDIM; kk += 32) {
    if (kk + 32 < CDIM)
      __builtin_prefetch(&Wqkv[(size_t)(kk + 32 + kb) * (3 * CDIM) + col], 0, 1);
    bf16x16 bmat;
#pragma unroll
    for (int e = 0; e < 16; ++e)
      bmat[e] = (__bf16)Wqkv[(size_t)(kk + kb + e) * (3 * CDIM) + col];
    bf16x16 amat[4];
#pragma unroll
    for (int mt = 0; mt < 4; ++mt) {
      const __bf16* ap = &xs[(mt * 16 + arowl) * CDIM + kk + ah];
      Pack8 lo, hi;
      lo.u = *(const uint4*)ap;
      hi.u = *(const uint4*)(ap + 8);
#pragma unroll
      for (int e = 0; e < 8; ++e) { amat[mt][e] = lo.h[e]; amat[mt][e + 8] = hi.h[e]; }
    }
#pragma unroll
    for (int mt = 0; mt < 4; ++mt)
      acc[mt] = wmma_bf16(amat[mt], bmat, acc[mt]);
  }

  const float bias  = bqkv[col];
  const int   which = col / CDIM;
  const int   cc    = col % CDIM;
  const int   hh    = cc / HDIM, dd = cc % HDIM;
  __bf16* dst = (which == 0) ? Qp : (which == 1) ? Kp : Vp;
  const float scl = (which == 0) ? 0.17677669529663687f : 1.0f;
  const int rb = 8 * (lane >> 4);
#pragma unroll
  for (int mt = 0; mt < 4; ++mt)
#pragma unroll
    for (int r = 0; r < 8; ++r) {
      int row = mt * 16 + rb + r;
      if (row < NTOK) {
        float v = (acc[mt][r] + bias) * scl;
        dst[(((size_t)b * HEADS + hh) * NTOK + row) * HDIM + dd] = (__bf16)v;
      }
    }
}

// ---------------------------------------------------------------------------
// Kernel B: fused attention per (b, h). Q/K/V tiles DMA'd into LDS by the
// Tensor Data Mover (one descriptor per wave 0..2), softmax with fused
// bias+mask table, both matmuls on WMMA bf16.
// ---------------------------------------------------------------------------
__global__ __launch_bounds__(128)
void attn_kernel(const __bf16* __restrict__ Qp,
                 const __bf16* __restrict__ Kp,
                 const __bf16* __restrict__ Vp,
                 const float*  __restrict__ biasP,
                 const float*  __restrict__ maskP,
                 __bf16* __restrict__ ctx) {
  __shared__ __bf16 Qs[NPAD * HDIM];
  __shared__ __bf16 Ks[NPAD * HDIM];
  __shared__ __bf16 Vs[NPAD * HDIM];
  __shared__ __bf16 Ps[NPAD * NPAD];
  __shared__ float  Ms[NPAD * NPAD];

  const int h = blockIdx.x, b = blockIdx.y;
  const int tid = threadIdx.x;
  const int wave = tid >> 5, lane = tid & 31;

  const size_t base = ((size_t)b * HEADS + h) * NTOK * HDIM;

#if USE_TDM
  if (wave < 3) {
    const __bf16* gsrc = (wave == 0) ? (Qp + base) : (wave == 1) ? (Kp + base) : (Vp + base);
    __bf16* ldst = (wave == 0) ? Qs : (wave == 1) ? Ks : Vs;
    // 49 rows x 32 bf16, contiguous (stride 32 elems)
    tdm_load_bf16_2d(gsrc, lds_offset_of(ldst), NTOK, HDIM, HDIM);
    __builtin_amdgcn_s_wait_tensorcnt(0);
  }
#else
  for (int i = tid; i < 196; i += 128) {
    ((uint4*)Qs)[i] = ((const uint4*)(Qp + base))[i];
    ((uint4*)Ks)[i] = ((const uint4*)(Kp + base))[i];
    ((uint4*)Vs)[i] = ((const uint4*)(Vp + base))[i];
  }
#endif
  const uint4 z4 = make_uint4(0, 0, 0, 0);
  for (int i = tid; i < 60; i += 128) {       // zero-pad rows 49..63
    ((uint4*)Qs)[196 + i] = z4;
    ((uint4*)Ks)[196 + i] = z4;
    ((uint4*)Vs)[196 + i] = z4;
  }
  // fused bias+mask 64x64 f32 into LDS
  const int grp = b & (NWIN - 1);
  const float* bsrc = biasP + h * NPAD * NPAD;
  const float* msrc = maskP + grp * NPAD * NPAD;
  for (int i = tid; i < NPAD * NPAD / 4; i += 128) {
    float4 a = *(const float4*)(bsrc + i * 4);
    float4 c = *(const float4*)(msrc + i * 4);
    *(float4*)&Ms[i * 4] =
        make_float4(a.x + c.x, a.y + c.y, a.z + c.z, a.w + c.w);
  }
  __syncthreads();

  const int m0    = wave * 16;
  const int mycol = lane & 15;
  const int rb    = 8 * (lane >> 4);
  const int ah    = (lane < 16) ? 0 : 16;
  const int kb    = (lane < 16) ? 0 : 16;

  bf16x16 aQ;
  {
    const __bf16* ap = &Qs[(m0 + mycol) * HDIM + ah];
    Pack8 lo, hi;
    lo.u = *(const uint4*)ap;
    hi.u = *(const uint4*)(ap + 8);
#pragma unroll
    for (int e = 0; e < 8; ++e) { aQ[e] = lo.h[e]; aQ[e + 8] = hi.h[e]; }
  }

  f32x8 s[4];
#pragma unroll
  for (int nt = 0; nt < 4; ++nt) {
    const __bf16* bp = &Ks[(nt * 16 + mycol) * HDIM + kb];
    Pack8 lo, hi;
    lo.u = *(const uint4*)bp;
    hi.u = *(const uint4*)(bp + 8);
    bf16x16 bK;
#pragma unroll
    for (int e = 0; e < 8; ++e) { bK[e] = lo.h[e]; bK[e + 8] = hi.h[e]; }
    s[nt] = wmma_bf16(aQ, bK, zero8());
  }

  float p[4][8];
#pragma unroll
  for (int r = 0; r < 8; ++r) {
    const int row = m0 + rb + r;
    float mx = -3.0e38f;
#pragma unroll
    for (int nt = 0; nt < 4; ++nt) {
      float v = s[nt][r] + Ms[row * NPAD + nt * 16 + mycol];
      p[nt][r] = v;
      mx = fmaxf(mx, v);
    }
    mx = fmaxf(mx, __shfl_xor(mx, 1, 32));
    mx = fmaxf(mx, __shfl_xor(mx, 2, 32));
    mx = fmaxf(mx, __shfl_xor(mx, 4, 32));
    mx = fmaxf(mx, __shfl_xor(mx, 8, 32));
    float sum = 0.f;
#pragma unroll
    for (int nt = 0; nt < 4; ++nt) {
      float e = __expf(p[nt][r] - mx);
      p[nt][r] = e;
      sum += e;
    }
    sum += __shfl_xor(sum, 1, 32);
    sum += __shfl_xor(sum, 2, 32);
    sum += __shfl_xor(sum, 4, 32);
    sum += __shfl_xor(sum, 8, 32);
    const float inv = 1.0f / sum;
#pragma unroll
    for (int nt = 0; nt < 4; ++nt)
      Ps[row * NPAD + nt * 16 + mycol] = (__bf16)(p[nt][r] * inv);
  }
  __syncthreads();

  f32x8 o[2] = {zero8(), zero8()};
#pragma unroll
  for (int kc = 0; kc < 2; ++kc) {
    const __bf16* ap = &Ps[(m0 + mycol) * NPAD + kc * 32 + ah];
    Pack8 lo, hi;
    lo.u = *(const uint4*)ap;
    hi.u = *(const uint4*)(ap + 8);
    bf16x16 aP;
#pragma unroll
    for (int e = 0; e < 8; ++e) { aP[e] = lo.h[e]; aP[e + 8] = hi.h[e]; }
#pragma unroll
    for (int nt2 = 0; nt2 < 2; ++nt2) {
      bf16x16 bV;
#pragma unroll
      for (int e = 0; e < 16; ++e)
        bV[e] = Vs[(kc * 32 + kb + e) * HDIM + nt2 * 16 + mycol];
      o[nt2] = wmma_bf16(aP, bV, o[nt2]);
    }
  }

#pragma unroll
  for (int nt2 = 0; nt2 < 2; ++nt2)
#pragma unroll
    for (int r = 0; r < 8; ++r) {
      const int row = m0 + rb + r;
      if (row < NTOK)
        ctx[((size_t)b * NTOK + row) * CDIM + h * HDIM + nt2 * 16 + mycol] =
            (__bf16)o[nt2][r];
    }
}

// ---------------------------------------------------------------------------
// Kernel C: out = ctx @ W_out + b_out. ctx tile (already bf16) is DMA'd to
// LDS with one TDM descriptor.
// ---------------------------------------------------------------------------
__global__ __launch_bounds__(256)
void out_gemm(const __bf16* __restrict__ ctx,
              const float* __restrict__ Wout,
              const float* __restrict__ bout,
              float* __restrict__ out) {
  __shared__ __bf16 cs[NPAD * CDIM];  // 48 KB
  const int b = blockIdx.y, cb = blockIdx.x;
  const int tid = threadIdx.x, wave = tid >> 5, lane = tid & 31;

#if USE_TDM
  if (wave == 0) {
    // 49*384 = 18816 contiguous bf16: one 1-row tile
    tdm_load_bf16_2d(ctx + (size_t)b * NTOK * CDIM, lds_offset_of(cs),
                     1u, (unsigned)(NTOK * CDIM), (unsigned)(NTOK * CDIM));
    __builtin_amdgcn_s_wait_tensorcnt(0);
  }
#else
  {
    const uint4* src = (const uint4*)(ctx + (size_t)b * NTOK * CDIM);
    for (int i = tid; i < 2352; i += 256) ((uint4*)cs)[i] = src[i];
  }
#endif
  for (int i = tid; i < 720; i += 256) ((uint4*)cs)[2352 + i] = make_uint4(0, 0, 0, 0);
  __syncthreads();

  const int colBase = cb * 128 + wave * 16;
  const int col     = colBase + (lane & 15);
  const int kb      = (lane < 16) ? 0 : 16;
  const int ah      = (lane < 16) ? 0 : 16;
  const int arowl   = lane & 15;

  f32x8 acc[4] = {zero8(), zero8(), zero8(), zero8()};
  for (int kk = 0; kk < CDIM; kk += 32) {
    if (kk + 32 < CDIM)
      __builtin_prefetch(&Wout[(size_t)(kk + 32 + kb) * CDIM + col], 0, 1);
    bf16x16 bmat;
#pragma unroll
    for (int e = 0; e < 16; ++e)
      bmat[e] = (__bf16)Wout[(size_t)(kk + kb + e) * CDIM + col];
    bf16x16 amat[4];
#pragma unroll
    for (int mt = 0; mt < 4; ++mt) {
      const __bf16* ap = &cs[(mt * 16 + arowl) * CDIM + kk + ah];
      Pack8 lo, hi;
      lo.u = *(const uint4*)ap;
      hi.u = *(const uint4*)(ap + 8);
#pragma unroll
      for (int e = 0; e < 8; ++e) { amat[mt][e] = lo.h[e]; amat[mt][e + 8] = hi.h[e]; }
    }
#pragma unroll
    for (int mt = 0; mt < 4; ++mt)
      acc[mt] = wmma_bf16(amat[mt], bmat, acc[mt]);
  }

  const float bias = bout[col];
  const int rb = 8 * (lane >> 4);
#pragma unroll
  for (int mt = 0; mt < 4; ++mt)
#pragma unroll
    for (int r = 0; r < 8; ++r) {
      int row = mt * 16 + rb + r;
      if (row < NTOK)
        out[((size_t)b * NTOK + row) * CDIM + col] = acc[mt][r] + bias;
    }
}

// ---------------------------------------------------------------------------
extern "C" void kernel_launch(void* const* d_in, const int* in_sizes, int n_in,
                              void* d_out, int out_size, void* d_ws, size_t ws_size,
                              hipStream_t stream) {
  const float* x      = (const float*)d_in[0];
  const float* mask   = (const float*)d_in[1];
  const float* Wqkv   = (const float*)d_in[2];
  const float* bqkv   = (const float*)d_in[3];
  const float* Wout   = (const float*)d_in[4];
  const float* bout   = (const float*)d_in[5];
  const float* rpb    = (const float*)d_in[6];
  const int*   relidx = (const int*)d_in[7];
  float*       out    = (float*)d_out;

  char* ws = (char*)d_ws;
  const size_t szQKV = (size_t)BWIN * HEADS * NTOK * HDIM * sizeof(__bf16);
  __bf16* Qp    = (__bf16*)ws;
  __bf16* Kp    = (__bf16*)(ws + szQKV);
  __bf16* Vp    = (__bf16*)(ws + 2 * szQKV);
  __bf16* ctx   = (__bf16*)(ws + 3 * szQKV);
  float*  biasP = (float*)(ws + 4 * szQKV);
  float*  maskP = biasP + HEADS * NPAD * NPAD;

  build_tables<<<(HEADS * NPAD * NPAD + NWIN * NPAD * NPAD) / 256, 256, 0, stream>>>(
      mask, rpb, relidx, biasP, maskP);
  qkv_gemm<<<dim3(9, BWIN), 256, 0, stream>>>(x, Wqkv, bqkv, Qp, Kp, Vp);
  attn_kernel<<<dim3(HEADS, BWIN), 128, 0, stream>>>(Qp, Kp, Vp, biasP, maskP, ctx);
  out_gemm<<<dim3(3, BWIN), 256, 0, stream>>>(ctx, Wout, bout, out);
}
